// LeroNet_ori_19447611916831
// MI455X (gfx1250) — matrix operand order repaired and checked by reading, not compiled, not run on hardware
//
#include <hip/hip_runtime.h>
#include <cstdint>
#include <cstddef>

// ---------------------------------------------------------------------------
// LeroNet tree-conv network for MI455X (gfx1250, wave32, WMMA bf16)
//
//   prep:   pack conv weights [O][C][3] -> bf16 [O][3C] (K = k*C + c)
//           transpose trees   [B][C][N] -> bf16 [B][N][C]
//   layerX: LDS <- norm+leaky(input); gather+GEMM via v_wmma_f32_16x16x32_bf16
//           (A staged in VGPRs per output-row block, B gathered from LDS,
//            branchless node-0 handling via 3 zero-prefixed index entries);
//           write raw bf16 out + per-tree (mean, 1/(std+eps)) stats
//   head:   maxpool (commutes with affine norm) + 64->32->1 MLP
// ---------------------------------------------------------------------------

#define B_TREES   512
#define N_PAD     512          // node axis incl. zero node
#define N_IDX     1533         // 3 * 511
#define LDS_PAD   8            // +8 halves per row: breaks bank-conflict stride

typedef __attribute__((ext_vector_type(16))) __bf16 v16bf;
typedef __attribute__((ext_vector_type(8)))  float  v8f;

union BF16x16 { uint4 q[2]; v16bf v; };

__device__ __forceinline__ unsigned short f2bf(float f) {
  unsigned int u = __float_as_uint(f);
  u += 0x7FFFu + ((u >> 16) & 1u);       // round-to-nearest-even
  return (unsigned short)(u >> 16);
}
__device__ __forceinline__ float bf2f(unsigned short h) {
  return __uint_as_float(((unsigned int)h) << 16);
}

// --------------------------- prep kernels ----------------------------------

__global__ void pack_conv_w(const float* __restrict__ W,
                            unsigned short* __restrict__ Wp, int O, int C) {
  int tot = O * C * 3;
  for (int i = blockIdx.x * blockDim.x + threadIdx.x; i < tot;
       i += gridDim.x * blockDim.x) {
    int o = i / (3 * C);
    int r = i % (3 * C);
    int k = r / C, c = r % C;
    Wp[(size_t)o * (3 * C) + k * C + c] = f2bf(W[((size_t)o * C + c) * 3 + k]);
  }
}

// trees [B][128][512] f32 -> X0 [B][512][128] bf16 (node-major for LDS gather)
__global__ void build_x0(const float* __restrict__ trees,
                         unsigned short* __restrict__ X0) {
  const size_t tot = (size_t)B_TREES * N_PAD * 128;
  for (size_t i = (size_t)blockIdx.x * blockDim.x + threadIdx.x; i < tot;
       i += (size_t)gridDim.x * blockDim.x) {
    int c = (int)(i % 128);
    size_t t = i / 128;
    int n = (int)(t % N_PAD);
    int b = (int)(t / N_PAD);
    X0[i] = f2bf(trees[((size_t)b * 128 + c) * N_PAD + n]);
  }
}

// --------------------------- tree-conv layer -------------------------------
// One workgroup per tree. 256 threads = 8 wave32.
// GEMM: out[COUT][512] = Wp[COUT][3*CIN] x G[3*CIN][512],
// G[k*CIN+c][n] = XT[idx[3(n-1)+k]][c]  (node 0: index table reads 0, output
// forced to raw 0 in the epilogue).
template <int CIN, int COUT, bool DO_NORM, bool DO_LEAKY>
__global__ __launch_bounds__(256) void tree_conv_layer(
    const unsigned short* __restrict__ inRaw,   // [B][512][CIN] bf16
    const int*            __restrict__ idx,     // [B][1533]
    const unsigned short* __restrict__ Wp,      // [COUT][3*CIN] bf16
    const float*          __restrict__ bias,    // [COUT]
    const float2*         __restrict__ statsIn, // per-tree (mean, inv) or null
    unsigned short*       __restrict__ outRaw,  // [B][512][COUT] bf16
    float2*               __restrict__ statsOut) {
  constexpr int ROW = CIN + LDS_PAD;
  extern __shared__ char smem[];
  unsigned short* XT  = (unsigned short*)smem;              // [512][ROW]
  int*   sidx3 = (int*)(smem + (size_t)N_PAD * ROW * 2);    // [3 + 1533]
  float* sred  = (float*)(sidx3 + 1540);                    // [16]

  const int b = blockIdx.x;

  // ---- load + fused norm/leaky into LDS (16B vector chunks) ----
  float mean = 0.f, inv = 1.f;
  if (DO_NORM) { float2 st = statsIn[b]; mean = st.x; inv = st.y; }
  const uint4* src = (const uint4*)(inRaw + (size_t)b * N_PAD * CIN);
  const int nchunk = N_PAD * CIN / 8;
  for (int t = threadIdx.x; t < nchunk; t += 256) {
    int n  = t / (CIN / 8);
    int c0 = (t % (CIN / 8)) * 8;
    uint4 q = src[t];
    if (DO_NORM) {
      unsigned short* hp = (unsigned short*)&q;
#pragma unroll
      for (int i = 0; i < 8; ++i) {
        float x = (bf2f(hp[i]) - mean) * inv;
        if (DO_LEAKY) x = (x >= 0.f) ? x : 0.01f * x;
        hp[i] = f2bf(x);
      }
    }
    *(uint4*)&XT[(size_t)n * ROW + c0] = q;
  }
  if (threadIdx.x < 3) sidx3[threadIdx.x] = 0;   // branchless node-0 gathers
  for (int t = threadIdx.x; t < N_IDX; t += 256)
    sidx3[3 + t] = idx[(size_t)b * N_IDX + t];
  __syncthreads();

  // ---- WMMA GEMM with on-the-fly gather ----
  const int lane = threadIdx.x & 31;
  const int wv   = threadIdx.x >> 5;
  const int lm   = lane & 15;
  const int hi   = lane >> 4;           // half-wave select
  constexpr int MT  = COUT / 16;
  constexpr int NT  = N_PAD / 16;
  constexpr int KT  = (3 * CIN) / 32;   // 32-K tiles
  constexpr int KST = 12;               // staged A tiles (384 K) per pass
  constexpr int KH  = KT / KST;         // 1 (CIN=128) or 2 (CIN=256)
  constexpr int NI  = NT / 8;           // node-tiles per wave per mt

  float sumv = 0.f, sumq = 0.f;

  for (int mt = 0; mt < MT; ++mt) {
    // gather indices for this wave's 4 node-tiles (one column per lane)
    int jj[NI][3];
#pragma unroll
    for (int i = 0; i < NI; ++i) {
      const int n = (wv + 8 * i) * 16 + lm;
#pragma unroll
      for (int k = 0; k < 3; ++k) jj[i][k] = sidx3[3 * n + k];
    }
    v8f acc[NI];
#pragma unroll
    for (int i = 0; i < NI; ++i) acc[i] = (v8f){};

    const unsigned short* wrow = Wp + (size_t)(mt * 16 + lm) * (3 * CIN);
#pragma unroll
    for (int kh = 0; kh < KH; ++kh) {
      // stage A tiles for this K-half in VGPRs (contiguous K order:
      // lanes0-15 K [0,16), lanes16-31 K [16,32))
      BF16x16 Areg[KST];
#pragma unroll
      for (int s = 0; s < KST; ++s) {
        const int B0 = (kh * KST + s) * 32 + hi * 16;
        Areg[s].q[0] = *(const uint4*)(wrow + B0);
        Areg[s].q[1] = *(const uint4*)(wrow + B0 + 8);
      }
#pragma unroll
      for (int i = 0; i < NI; ++i) {
#pragma unroll
        for (int s = 0; s < KST; ++s) {
          const int kt  = kh * KST + s;
          const int Kb  = kt * 32 + hi * 16;
          const int k   = Kb / CIN;          // compile-time per s
          const int c16 = Kb % CIN;
          const unsigned short* xp = &XT[(size_t)jj[i][k] * ROW + c16];
          BF16x16 Bv;
          Bv.q[0] = *(const uint4*)xp;
          Bv.q[1] = *(const uint4*)(xp + 8);
          acc[i] = __builtin_amdgcn_wmma_f32_16x16x32_bf16(
              false, Areg[s].v, false, Bv.v, (short)0, acc[i], false, false);
        }
      }
    }
    // epilogue: bias, zero node-0 column, stats, bf16 store (8 ch / lane)
    const float4* bp = (const float4*)(bias + mt * 16 + hi * 8);
    const float4 b0 = bp[0], b1 = bp[1];
    const float bb[8] = {b0.x, b0.y, b0.z, b0.w, b1.x, b1.y, b1.z, b1.w};
#pragma unroll
    for (int i = 0; i < NI; ++i) {
      const int n = (wv + 8 * i) * 16 + lm;
      unsigned short ob[8];
#pragma unroll
      for (int r = 0; r < 8; ++r) {
        float val = (n == 0) ? 0.f : (acc[i][r] + bb[r]);
        sumv += val;
        sumq += val * val;
        ob[r] = f2bf(val);
      }
      *(uint4*)&outRaw[((size_t)b * N_PAD + n) * COUT + mt * 16 + hi * 8] =
          *(uint4*)ob;
    }
  }

  // ---- per-tree mean / inv-std reduction ----
#pragma unroll
  for (int o = 16; o > 0; o >>= 1) {
    sumv += __shfl_xor(sumv, o, 32);
    sumq += __shfl_xor(sumq, o, 32);
  }
  if (lane == 0) { sred[wv] = sumv; sred[8 + wv] = sumq; }
  __syncthreads();
  if (threadIdx.x == 0) {
    float S = 0.f, Q = 0.f;
#pragma unroll
    for (int i = 0; i < 8; ++i) { S += sred[i]; Q += sred[8 + i]; }
    const float ne  = (float)(COUT * N_PAD);
    const float mu  = S / ne;
    const float var = (Q - ne * mu * mu) / (ne - 1.f);
    statsOut[b] = make_float2(mu, 1.f / (sqrtf(var) + 1e-5f));
  }
}

// --------------------------- head: maxpool + MLP ---------------------------
__global__ __launch_bounds__(256) void final_head(
    const unsigned short* __restrict__ raw3,   // [B][512][64] bf16 (pre-norm)
    const float2* __restrict__ stats3,
    const float* __restrict__ Wf1, const float* __restrict__ bf1,
    const float* __restrict__ Wf2, const float* __restrict__ bf2,
    float* __restrict__ out) {
  __shared__ float part[4][64];
  __shared__ float pool[64];
  __shared__ float hbuf[32];
  const int b = blockIdx.x;
  const int c = threadIdx.x & 63;
  const int g = threadIdx.x >> 6;
  float m = -3.4e38f;
  for (int n = g; n < N_PAD; n += 4)
    m = fmaxf(m, bf2f(raw3[((size_t)b * N_PAD + n) * 64 + c]));
  part[g][c] = m;
  __syncthreads();
  if (threadIdx.x < 64) {
    float mm = fmaxf(fmaxf(part[0][c], part[1][c]),
                     fmaxf(part[2][c], part[3][c]));
    float2 st = stats3[b];
    pool[c] = (mm - st.x) * st.y;      // norm commutes with max (inv > 0)
  }
  __syncthreads();
  if (threadIdx.x < 32) {
    float h = bf1[threadIdx.x];
#pragma unroll
    for (int i = 0; i < 64; ++i) h += pool[i] * Wf1[threadIdx.x * 64 + i];
    hbuf[threadIdx.x] = (h >= 0.f) ? h : 0.01f * h;
  }
  __syncthreads();
  if (threadIdx.x == 0) {
    float o = bf2[0];
#pragma unroll
    for (int j = 0; j < 32; ++j) o += hbuf[j] * Wf2[j];
    out[b] = o;
  }
}

// --------------------------- host launcher ---------------------------------

static inline size_t align256(size_t x) { return (x + 255) & ~(size_t)255; }

extern "C" void kernel_launch(void* const* d_in, const int* in_sizes, int n_in,
                              void* d_out, int out_size, void* d_ws,
                              size_t ws_size, hipStream_t stream) {
  (void)in_sizes; (void)n_in; (void)out_size; (void)ws_size;
  const float* trees = (const float*)d_in[0];
  const int*   idx   = (const int*)d_in[1];     // harness delivers ints as i32
  const float* W1 = (const float*)d_in[2];  const float* b1 = (const float*)d_in[3];
  const float* W2 = (const float*)d_in[4];  const float* b2 = (const float*)d_in[5];
  const float* W3 = (const float*)d_in[6];  const float* b3 = (const float*)d_in[7];
  const float* Wf1 = (const float*)d_in[8]; const float* bf1 = (const float*)d_in[9];
  const float* Wf2 = (const float*)d_in[10];const float* bf2 = (const float*)d_in[11];
  float* out = (float*)d_out;

  // workspace carve-up (bufA overlays X0/raw2, bufB overlays raw1/raw3)
  char* ws = (char*)d_ws;
  size_t off = 0;
  unsigned short* Wp1 = (unsigned short*)(ws + off); off = align256(off + (size_t)256 * 384 * 2);
  unsigned short* Wp2 = (unsigned short*)(ws + off); off = align256(off + (size_t)128 * 768 * 2);
  unsigned short* Wp3 = (unsigned short*)(ws + off); off = align256(off + (size_t)64 * 384 * 2);
  float2* st1 = (float2*)(ws + off); off = align256(off + B_TREES * sizeof(float2));
  float2* st2 = (float2*)(ws + off); off = align256(off + B_TREES * sizeof(float2));
  float2* st3 = (float2*)(ws + off); off = align256(off + B_TREES * sizeof(float2));
  unsigned short* bufA = (unsigned short*)(ws + off);            // 67 MB
  off = align256(off + (size_t)B_TREES * N_PAD * 256 * 2 / 2);   // max(X0, raw2)
  unsigned short* bufB = (unsigned short*)(ws + off);            // 134 MB
  unsigned short* X0   = bufA;   // [B][512][128]
  unsigned short* raw1 = bufB;   // [B][512][256]
  unsigned short* raw2 = bufA;   // [B][512][128]
  unsigned short* raw3 = bufB;   // [B][512][64]

  constexpr size_t LDS_S = (size_t)N_PAD * (128 + LDS_PAD) * 2 + 1540 * 4 + 64;
  constexpr size_t LDS_L = (size_t)N_PAD * (256 + LDS_PAD) * 2 + 1540 * 4 + 64;

  auto k1 = tree_conv_layer<128, 256, false, false>;
  auto k2 = tree_conv_layer<256, 128, true,  true >;
  auto k3 = tree_conv_layer<128, 64,  true,  true >;
  // allow >64KB dynamic LDS (idempotent; not a stream op -> capture-safe)
  (void)hipFuncSetAttribute((const void*)k1, hipFuncAttributeMaxDynamicSharedMemorySize, (int)LDS_S);
  (void)hipFuncSetAttribute((const void*)k2, hipFuncAttributeMaxDynamicSharedMemorySize, (int)LDS_L);
  (void)hipFuncSetAttribute((const void*)k3, hipFuncAttributeMaxDynamicSharedMemorySize, (int)LDS_S);

  pack_conv_w<<<384, 256, 0, stream>>>(W1, Wp1, 256, 128);
  pack_conv_w<<<384, 256, 0, stream>>>(W2, Wp2, 128, 256);
  pack_conv_w<<<96,  256, 0, stream>>>(W3, Wp3, 64, 128);
  build_x0<<<4096, 256, 0, stream>>>(trees, X0);

  k1<<<B_TREES, 256, LDS_S, stream>>>(X0,   idx, Wp1, b1, nullptr, raw1, st1);
  k2<<<B_TREES, 256, LDS_L, stream>>>(raw1, idx, Wp2, b2, st1,     raw2, st2);
  k3<<<B_TREES, 256, LDS_S, stream>>>(raw2, idx, Wp3, b3, st2,     raw3, st3);

  final_head<<<B_TREES, 256, 0, stream>>>(raw3, st3, Wf1, bf1, Wf2, bf2, out);
}
